// MultiheadAttention_47184510713977
// MI455X (gfx1250) — compile-verified, matrix operand order checked by reading, not run on hardware
//
#include <hip/hip_runtime.h>
#include <hip/hip_bf16.h>

// MI455X (gfx1250) multi-head attention forward, bf16 WMMA pipeline.
// BS=4, N_CTX=2048, WIDTH=1024, HEADS=16, HEAD_DIM=64.

#define DEVINL __device__ __forceinline__

typedef __attribute__((ext_vector_type(16))) __bf16 bf16x16;
typedef __attribute__((ext_vector_type(8)))  float  f32x8;
typedef __attribute__((ext_vector_type(4)))  int    v4i_;

#define BS_   4
#define NCTX_ 2048
#define WIDTH_ 1024
#define HEADS_ 16
#define HDIM_ 64
#define SCALE_ 0.35355339059327373f   // 1/sqrt(sqrt(64))

#define GLOBAL_AS __attribute__((address_space(1)))
#define LDS_AS    __attribute__((address_space(3)))

// ---- CDNA5 async global->LDS staging (ASYNCcnt), guarded --------------
#if defined(__gfx1250__) && __has_builtin(__builtin_amdgcn_global_load_async_to_lds_b128)
#define HAS_ASYNC_LDS 1
DEVINL void async_b128(const void* g, void* l) {
  // signature: (v4i AS1* src, v4i AS3* dst, imm offset, imm cpol)
  __builtin_amdgcn_global_load_async_to_lds_b128(
      (GLOBAL_AS v4i_*)(g), (LDS_AS v4i_*)(l), 0, 0);
}
DEVINL void wait_async0() {
#if __has_builtin(__builtin_amdgcn_s_wait_asynccnt)
  __builtin_amdgcn_s_wait_asynccnt(0);
#else
  asm volatile("s_wait_asynccnt 0x0" ::: "memory");
#endif
}
#else
#define HAS_ASYNC_LDS 0
#endif

// ---- DPP half-wave reductions (VALU pipe, no LDS traffic) -------------
#if defined(__gfx1250__) && __has_builtin(__builtin_amdgcn_mov_dpp8) && \
    __has_builtin(__builtin_amdgcn_update_dpp)
#define HAS_DPP 1
constexpr unsigned dpp8_sel_xor(int m) {
  unsigned v = 0;
  for (int i = 0; i < 8; ++i) v |= (unsigned)((i ^ m) & 7) << (3 * i);
  return v;
}
template <int M>
DEVINL float dpp8_xor_f(float x) {
  constexpr unsigned sel = dpp8_sel_xor(M);
  return __int_as_float(__builtin_amdgcn_mov_dpp8(__float_as_int(x), sel));
}
DEVINL float dpp16_ror8_f(float x) {  // lane i <- lane (i^8) within each 16-row
  int i = __float_as_int(x);
  return __int_as_float(__builtin_amdgcn_update_dpp(i, i, 0x128, 0xf, 0xf, false));
}
DEVINL float red16_max(float v) {
  v = fmaxf(v, dpp8_xor_f<1>(v));
  v = fmaxf(v, dpp8_xor_f<2>(v));
  v = fmaxf(v, dpp8_xor_f<4>(v));
  v = fmaxf(v, dpp16_ror8_f(v));
  return v;
}
DEVINL float red16_sum(float v) {
  v += dpp8_xor_f<1>(v);
  v += dpp8_xor_f<2>(v);
  v += dpp8_xor_f<4>(v);
  v += dpp16_ror8_f(v);
  return v;
}
#else
#define HAS_DPP 0
DEVINL float red16_max(float v) {
  v = fmaxf(v, __shfl_xor(v, 1));
  v = fmaxf(v, __shfl_xor(v, 2));
  v = fmaxf(v, __shfl_xor(v, 4));
  v = fmaxf(v, __shfl_xor(v, 8));
  return v;
}
DEVINL float red16_sum(float v) {
  v += __shfl_xor(v, 1);
  v += __shfl_xor(v, 2);
  v += __shfl_xor(v, 4);
  v += __shfl_xor(v, 8);
  return v;
}
#endif

DEVINL f32x8 wmma_bf16(bf16x16 a, bf16x16 b, f32x8 c) {
  return __builtin_amdgcn_wmma_f32_16x16x32_bf16(
      false, a, false, b, (short)0, c, false, false);
}

// 16x32 bf16 fragment (ISA 7.12.2 layout) from row-major [row][k] buffer.
// Element e of lane L: row = row0 + (L&15), K = k0 + 16*(e>>3) + 8*(L>>4) + (e&7)
// => two contiguous 16B loads per lane. stride must be a multiple of 8.
DEVINL bf16x16 load_frag(const __bf16* __restrict__ base, int row0, int k0, int stride) {
  const int lane = threadIdx.x & 31;
  const int r = lane & 15;
  const int h = lane >> 4;
  const __bf16* p = base + (size_t)(row0 + r) * stride + k0 + h * 8;
  union { uint4 u[2]; bf16x16 v; } f;
  f.u[0] = *(const uint4*)(p);
  f.u[1] = *(const uint4*)(p + 16);
  return f.v;
}

// ---------------------------------------------------------------- convert
__global__ __launch_bounds__(256) void cvt_f32_bf16(const float* __restrict__ in,
                                                    __bf16* __restrict__ out, int n4) {
  int i = blockIdx.x * 256 + threadIdx.x;
  if (i < n4) {
    float4 v = ((const float4*)in)[i];
    union { __bf16 e[4]; uint2 u; } o;
    o.e[0] = (__bf16)v.x; o.e[1] = (__bf16)v.y;
    o.e[2] = (__bf16)v.z; o.e[3] = (__bf16)v.w;
    ((uint2*)out)[i] = o.u;
  }
}

// in: [K][N] f32  ->  outT: [N][K] bf16 (one-time weight transpose)
__global__ __launch_bounds__(256) void cvt_f32_bf16_T(const float* __restrict__ in,
                                                      __bf16* __restrict__ outT,
                                                      int K, int N) {
  int i = blockIdx.x * 256 + threadIdx.x;
  int k = i / (N >> 2);
  int n4 = i - k * (N >> 2);
  if (k < K) {
    float4 v = ((const float4*)(in + (size_t)k * N))[n4];
    int n = n4 * 4;
    outT[(size_t)(n + 0) * K + k] = (__bf16)v.x;
    outT[(size_t)(n + 1) * K + k] = (__bf16)v.y;
    outT[(size_t)(n + 2) * K + k] = (__bf16)v.z;
    outT[(size_t)(n + 3) * K + k] = (__bf16)v.w;
  }
}

// ---------------------------------------------------------------- GEMM
// C[M,N] = A[M,K] * Bt[N,K]^T (+bias). 128x128x32 tile, 256 thr = 8 waves
// (2Mx4N); each wave owns 64x32 = 4x2 WMMA tiles. Both LDS tiles are pure
// contiguous b128 copies (async-to-LDS when available).
// MODE 0: QKV epilogue -> Q*scale,K*scale [B,H,N,64]; V transposed [B,H,64,N]
// MODE 1: proj epilogue -> f32 out + bias
template <int MODE>
__global__ __launch_bounds__(256) void gemm_bf16_kernel(
    const __bf16* __restrict__ A, const __bf16* __restrict__ Bt,
    const float* __restrict__ bias, float* __restrict__ outF,
    __bf16* __restrict__ Qb, __bf16* __restrict__ Kb, __bf16* __restrict__ Vb,
    int M, int N, int K) {
  __shared__ __bf16 sA[128 * 40];   // [m][k], stride 40 (16B-aligned rows)
  __shared__ __bf16 sB[128 * 40];   // [n][k]

  const int tid = threadIdx.x;
  const int lane = tid & 31;
  const int wid = tid >> 5;
  const int lr = lane & 15;
  const int lh = lane >> 4;
  const int wave_m = wid >> 2;
  const int wave_n = wid & 3;
  const int m0 = blockIdx.y * 128;
  const int n0 = blockIdx.x * 128;

  f32x8 zero = {0.f, 0.f, 0.f, 0.f, 0.f, 0.f, 0.f, 0.f};
  f32x8 acc[4][2];
#pragma unroll
  for (int i = 0; i < 4; ++i)
#pragma unroll
    for (int j = 0; j < 2; ++j) acc[i][j] = zero;

  const int kTiles = K >> 5;
  for (int kk = 0; kk < kTiles; ++kk) {
    __syncthreads();
    // 128 rows x 32 bf16 per tile = 512 x b128; 2 per thread per matrix
#pragma unroll
    for (int i = 0; i < 2; ++i) {
      int u = tid * 2 + i;
      int row = u >> 2, c4 = u & 3;
      const __bf16* ag = A + (size_t)(m0 + row) * K + kk * 32 + c4 * 8;
      const __bf16* bg = Bt + (size_t)(n0 + row) * K + kk * 32 + c4 * 8;
      __bf16* al = sA + row * 40 + c4 * 8;
      __bf16* bl = sB + row * 40 + c4 * 8;
#if HAS_ASYNC_LDS
      async_b128(ag, al);
      async_b128(bg, bl);
#else
      *(uint4*)al = *(const uint4*)ag;
      *(uint4*)bl = *(const uint4*)bg;
#endif
    }
    if (kk + 1 < kTiles) {  // global_prefetch_b8 for next tiles
      __builtin_prefetch(A + (size_t)(m0 + (tid >> 1)) * K + (kk + 1) * 32, 0, 3);
      __builtin_prefetch(Bt + (size_t)(n0 + (tid >> 1)) * K + (kk + 1) * 32, 0, 3);
    }
#if HAS_ASYNC_LDS
    wait_async0();
#endif
    __syncthreads();

    bf16x16 bfr[2];
#pragma unroll
    for (int tn = 0; tn < 2; ++tn)
      bfr[tn] = load_frag(sB, wave_n * 32 + tn * 16, 0, 40);
#pragma unroll
    for (int tm = 0; tm < 4; ++tm) {
      bf16x16 afr = load_frag(sA, wave_m * 64 + tm * 16, 0, 40);
#pragma unroll
      for (int tn = 0; tn < 2; ++tn)
        acc[tm][tn] = wmma_bf16(afr, bfr[tn], acc[tm][tn]);
    }
  }

  // Epilogue. C layout: VGPR r holds row (r + 8*lh), column = lane&15.
#pragma unroll
  for (int tm = 0; tm < 4; ++tm) {
#pragma unroll
    for (int tn = 0; tn < 2; ++tn) {
      int c = n0 + wave_n * 32 + tn * 16 + lr;
      float bv = bias[c];
      if (MODE == 0) {
        int hh = c / 192, r2 = c % 192;
        int which = r2 >> 6, d = r2 & 63;
#pragma unroll
        for (int r = 0; r < 8; ++r) {
          int m = m0 + wave_m * 64 + tm * 16 + r + 8 * lh;
          int bidx = m >> 11, t = m & (NCTX_ - 1);
          float val = acc[tm][tn][r] + bv;
          size_t bhh = (size_t)(bidx * HEADS_ + hh);
          if (which == 0)
            Qb[(bhh * NCTX_ + t) * HDIM_ + d] = (__bf16)(val * SCALE_);
          else if (which == 1)
            Kb[(bhh * NCTX_ + t) * HDIM_ + d] = (__bf16)(val * SCALE_);
          else  // V stored transposed: [B,H,D,N]
            Vb[(bhh * HDIM_ + d) * NCTX_ + t] = (__bf16)val;
        }
      } else {
#pragma unroll
        for (int r = 0; r < 8; ++r) {
          int m = m0 + wave_m * 64 + tm * 16 + r + 8 * lh;
          outF[(size_t)m * N + c] = acc[tm][tn][r] + bv;
        }
      }
    }
  }
}

// ---------------------------------------------------------------- attention
// Grid: (NCTX/128, BS*HEADS). 8 waves; wave w owns 16 query rows. Flash
// attention over key blocks of 64; K and V^T tiles staged with contiguous
// b128 copies (async when available).
__global__ __launch_bounds__(256) void attn_kernel(
    const __bf16* __restrict__ Qb, const __bf16* __restrict__ Kb,
    const __bf16* __restrict__ Vb, const unsigned char* __restrict__ mask,
    __bf16* __restrict__ attnb) {
  __shared__ __bf16 sK[64 * 72];        // [key][dim]
  __shared__ __bf16 sVt[64 * 72];       // [dim][key]
  __shared__ __bf16 sP[8 * 16 * 72];    // per-wave P strip [16][64]
  __shared__ float smask[64];

  const int tid = threadIdx.x;
  const int lane = tid & 31;
  const int wid = tid >> 5;
  const int lr = lane & 15, lh = lane >> 4;
  const int bh = blockIdx.y;            // b*HEADS + h
  const int b = bh >> 4, h = bh & 15;
  const int q0 = blockIdx.x * 128;

  const __bf16* Qrow = Qb + ((size_t)bh * NCTX_ + q0 + wid * 16) * HDIM_;
  bf16x16 qf[2];
  qf[0] = load_frag(Qrow, 0, 0, HDIM_);
  qf[1] = load_frag(Qrow, 0, 32, HDIM_);

  f32x8 zero = {0.f, 0.f, 0.f, 0.f, 0.f, 0.f, 0.f, 0.f};
  f32x8 o[4];
#pragma unroll
  for (int ot = 0; ot < 4; ++ot) o[ot] = zero;
  float mprev[8], lsum[8];
#pragma unroll
  for (int r = 0; r < 8; ++r) { mprev[r] = -3.0e38f; lsum[r] = 0.f; }

  const __bf16* Kbh = Kb + (size_t)bh * NCTX_ * HDIM_;
  const __bf16* Vbh = Vb + (size_t)bh * HDIM_ * NCTX_;  // [dim][t]
  __bf16* myP = sP + wid * 16 * 72;

  for (int kb = 0; kb < NCTX_ / 64; ++kb) {
    __syncthreads();
#pragma unroll
    for (int i = 0; i < 2; ++i) {
      int u = tid * 2 + i;
      int row = u >> 3, c4 = u & 7;
      const __bf16* kg = Kbh + (size_t)(kb * 64 + row) * HDIM_ + c4 * 8;
      const __bf16* vg = Vbh + (size_t)row * NCTX_ + kb * 64 + c4 * 8;
      __bf16* kl = sK + row * 72 + c4 * 8;
      __bf16* vl = sVt + row * 72 + c4 * 8;
#if HAS_ASYNC_LDS
      async_b128(kg, kl);
      async_b128(vg, vl);
#else
      *(uint4*)kl = *(const uint4*)kg;
      *(uint4*)vl = *(const uint4*)vg;
#endif
    }
    if (tid < 64)
      smask[tid] = mask[b * NCTX_ + kb * 64 + tid] ? 0.0f : -3.0e38f;
#if HAS_ASYNC_LDS
    wait_async0();
#endif
    __syncthreads();

    // S = Q * K^T (16 rows x 64 keys per wave), mask folded in additively
    f32x8 s[4];
#pragma unroll
    for (int ct = 0; ct < 4; ++ct) {
      f32x8 c = zero;
      c = wmma_bf16(qf[0], load_frag(sK, ct * 16, 0, 72), c);
      c = wmma_bf16(qf[1], load_frag(sK, ct * 16, 32, 72), c);
      float madd = smask[ct * 16 + lr];
#pragma unroll
      for (int r = 0; r < 8; ++r) c[r] += madd;
      s[ct] = c;
    }

    // Online softmax; row (r + 8*lh) spans 16 lanes -> DPP reductions
#pragma unroll
    for (int r = 0; r < 8; ++r) {
      float mx = fmaxf(fmaxf(s[0][r], s[1][r]), fmaxf(s[2][r], s[3][r]));
      mx = red16_max(mx);
      float mnew = fmaxf(mprev[r], mx);
      float corr = __expf(mprev[r] - mnew);
      float part = 0.f;
#pragma unroll
      for (int ct = 0; ct < 4; ++ct) {
        float p = __expf(s[ct][r] - mnew);
        s[ct][r] = p;
        part += p;
      }
      part = red16_sum(part);
      lsum[r] = lsum[r] * corr + part;
#pragma unroll
      for (int ot = 0; ot < 4; ++ot) o[ot][r] *= corr;
      mprev[r] = mnew;
    }

    // C-layout -> A-layout for P via wave-private LDS (in-order per wave)
#pragma unroll
    for (int ct = 0; ct < 4; ++ct)
#pragma unroll
      for (int r = 0; r < 8; ++r)
        myP[(r + 8 * lh) * 72 + ct * 16 + lr] = (__bf16)s[ct][r];

    // O += P * V
#pragma unroll
    for (int g = 0; g < 2; ++g) {
      bf16x16 pf = load_frag(myP, 0, g * 32, 72);
#pragma unroll
      for (int ot = 0; ot < 4; ++ot)
        o[ot] = wmma_bf16(pf, load_frag(sVt, ot * 16, g * 32, 72), o[ot]);
    }
  }

  float invl[8];
#pragma unroll
  for (int r = 0; r < 8; ++r) invl[r] = 1.0f / lsum[r];
#pragma unroll
  for (int ot = 0; ot < 4; ++ot) {
    int d = h * HDIM_ + ot * 16 + lr;
#pragma unroll
    for (int r = 0; r < 8; ++r) {
      int tq = q0 + wid * 16 + r + 8 * lh;
      attnb[((size_t)b * NCTX_ + tq) * WIDTH_ + d] = (__bf16)(o[ot][r] * invl[r]);
    }
  }
}

// ---------------------------------------------------------------- launch
extern "C" void kernel_launch(void* const* d_in, const int* in_sizes, int n_in,
                              void* d_out, int out_size, void* d_ws, size_t ws_size,
                              hipStream_t stream) {
  const float* x     = (const float*)d_in[0];
  const unsigned char* mask = (const unsigned char*)d_in[1];  // jnp bool = 1B
  const float* Wqkv  = (const float*)d_in[2];
  const float* bqkv  = (const float*)d_in[3];
  const float* Wproj = (const float*)d_in[4];
  const float* bproj = (const float*)d_in[5];
  float* out = (float*)d_out;

  char* p = (char*)d_ws;
  const size_t TOK = (size_t)BS_ * NCTX_;                   // 8192
  const size_t HEL = (size_t)BS_ * HEADS_ * NCTX_ * HDIM_;  // per Q/K/V
  __bf16* xb     = (__bf16*)p; p += TOK * WIDTH_ * 2;
  __bf16* wqkvT  = (__bf16*)p; p += (size_t)WIDTH_ * 3 * WIDTH_ * 2;  // [3W][W]
  __bf16* wprojT = (__bf16*)p; p += (size_t)WIDTH_ * WIDTH_ * 2;      // [W][W]
  __bf16* Qb     = (__bf16*)p; p += HEL * 2;
  __bf16* Kb     = (__bf16*)p; p += HEL * 2;
  __bf16* Vb     = (__bf16*)p; p += HEL * 2;   // transposed [B,H,D,N]
  __bf16* attnb  = (__bf16*)p; p += TOK * WIDTH_ * 2;

  cvt_f32_bf16<<<(TOK * WIDTH_ / 4 + 255) / 256, 256, 0, stream>>>(
      x, xb, TOK * WIDTH_ / 4);
  cvt_f32_bf16_T<<<(3 * WIDTH_ * WIDTH_ / 4 + 255) / 256, 256, 0, stream>>>(
      Wqkv, wqkvT, WIDTH_, 3 * WIDTH_);
  cvt_f32_bf16_T<<<(WIDTH_ * WIDTH_ / 4 + 255) / 256, 256, 0, stream>>>(
      Wproj, wprojT, WIDTH_, WIDTH_);

  gemm_bf16_kernel<0><<<dim3(3 * WIDTH_ / 128, TOK / 128), 256, 0, stream>>>(
      xb, wqkvT, bqkv, nullptr, Qb, Kb, Vb, (int)TOK, 3 * WIDTH_, WIDTH_);

  attn_kernel<<<dim3(NCTX_ / 128, BS_ * HEADS_), 256, 0, stream>>>(
      Qb, Kb, Vb, mask, attnb);

  gemm_bf16_kernel<1><<<dim3(WIDTH_ / 128, TOK / 128), 256, 0, stream>>>(
      attnb, wprojT, bproj, out, nullptr, nullptr, nullptr,
      (int)TOK, WIDTH_, WIDTH_);
}
